// MultiHeadAttention_57509612094070
// MI455X (gfx1250) — matrix operand order repaired and checked by reading, not compiled, not run on hardware
//
#include <hip/hip_runtime.h>
#include <hip/hip_bf16.h>

// ---------------------------------------------------------------------------
// MHA forward on gfx1250 via V_WMMA_F32_16X16X32_F16.
// Pipeline: f32->f16 converts, 3 WMMA projection GEMMs, flash-style WMMA
// attention with online softmax, WMMA output GEMM (+bias) in f32.
// Round 2: softmax row-sum via ones-WMMA (removes 32 ds_bpermute/iter),
// row-max via DPP16 butterfly (removes the rest), SCALE folded into q frags.
// ---------------------------------------------------------------------------

typedef __attribute__((ext_vector_type(16))) _Float16 v16h;
typedef __attribute__((ext_vector_type(8)))  _Float16 v8h;
typedef __attribute__((ext_vector_type(8)))  float    v8f;

#define DEV __device__ __forceinline__

// A-operand fragment (16xK tile, rows per lane):
//   lane L holds row (L&15); K elements at {hlf*8 .. +7} and {16+hlf*8 .. +7}
DEV v16h load_frag_a(const _Float16* p, int ld) {
  const int lane = threadIdx.x & 31;
  const int row = lane & 15, hlf = lane >> 4;
  const _Float16* q = p + (size_t)row * ld + hlf * 8;
  v8h lo = *(const v8h*)(q);
  v8h hi = *(const v8h*)(q + 16);
  return __builtin_shufflevector(lo, hi, 0,1,2,3,4,5,6,7,8,9,10,11,12,13,14,15);
}

// B-operand fragment (Kx16 tile, columns per lane):
//   lane L holds column (L&15); K elements at {hlf*8 .. +7} and {16+hlf*8 .. +7}
DEV v16h load_frag_b(const _Float16* p, int ld) {
  const int lane = threadIdx.x & 31;
  const int col = lane & 15, hlf = lane >> 4;
  const _Float16* q = p + (size_t)col * ld + hlf * 8;
  v8h lo = *(const v8h*)(q);
  v8h hi = *(const v8h*)(q + 16);
  return __builtin_shufflevector(lo, hi, 0,1,2,3,4,5,6,7,8,9,10,11,12,13,14,15);
}

DEV v8f wmma_f16(v16h a, v16h b, v8f c) {
  return __builtin_amdgcn_wmma_f32_16x16x32_f16(false, a, false, b,
                                                (short)0, c, false, false);
}

// Max all-reduce + broadcast over a 16-lane half using a DPP16 butterfly.
// Stage masks {1,2,7,15} span bits[3:0] -> full reduction, zero LDS traffic.
DEV float dpp_bcast_max16(float x) {
  int t;
  t = __builtin_amdgcn_update_dpp(0, __float_as_int(x), 0xB1,  0xF, 0xF, true); // quad_perm xor1
  x = fmaxf(x, __int_as_float(t));
  t = __builtin_amdgcn_update_dpp(0, __float_as_int(x), 0x4E,  0xF, 0xF, true); // quad_perm xor2
  x = fmaxf(x, __int_as_float(t));
  t = __builtin_amdgcn_update_dpp(0, __float_as_int(x), 0x141, 0xF, 0xF, true); // row_half_mirror
  x = fmaxf(x, __int_as_float(t));
  t = __builtin_amdgcn_update_dpp(0, __float_as_int(x), 0x140, 0xF, 0xF, true); // row_mirror
  x = fmaxf(x, __int_as_float(t));
  return x;
}

// ---------------------------------------------------------------------------
// Elementwise f32 -> f16 convert
// ---------------------------------------------------------------------------
__global__ __launch_bounds__(256)
void cvt_f32_f16_kernel(const float* __restrict__ in, _Float16* __restrict__ out, int n) {
  for (int i = blockIdx.x * blockDim.x + threadIdx.x; i < n; i += gridDim.x * blockDim.x)
    out[i] = (_Float16)in[i];
}

// W (rows x cols, f32) -> Wt (cols x rows, f16)
__global__ __launch_bounds__(256)
void cvt_transpose_kernel(const float* __restrict__ in, _Float16* __restrict__ out,
                          int rows, int cols) {
  int idx = blockIdx.x * blockDim.x + threadIdx.x;
  if (idx >= rows * cols) return;
  int k = idx / cols, n = idx - k * cols;
  out[(size_t)n * rows + k] = (_Float16)in[idx];
}

// ---------------------------------------------------------------------------
// WMMA GEMM: C(MxN) = A(MxK) * Bt(NxK)^T ; f32 accumulate.
// 256 threads (8 waves); block tile 128x128; wave tile 32x64 (2x4 WMMA accums)
// ---------------------------------------------------------------------------
enum { EPI_F16 = 0, EPI_VT = 1, EPI_F32_BIAS = 2 };

template <int EPI>
__global__ __launch_bounds__(256)
void gemm_wmma_kernel(const _Float16* __restrict__ A, const _Float16* __restrict__ Bt,
                      void* __restrict__ Out, const float* __restrict__ bias,
                      int M, int Nd, int K) {
  const int w   = threadIdx.x >> 5;
  const int wm  = w >> 1, wn = w & 1;
  const int m0  = blockIdx.y * 128 + wm * 32;
  const int n0  = blockIdx.x * 128 + wn * 64;
  const int lane = threadIdx.x & 31;
  const int hlf = lane >> 4, col = lane & 15;

  v8f acc[2][4];
#pragma unroll
  for (int i = 0; i < 2; ++i)
#pragma unroll
    for (int j = 0; j < 4; ++j) acc[i][j] = (v8f){};

  for (int k0 = 0; k0 < K; k0 += 32) {
    if (k0 + 32 < K) {  // prefetch next K tile -> global_prefetch_b8
      __builtin_prefetch((const void*)(A  + (size_t)(m0 + lane)      * K + k0 + 32), 0, 3);
      __builtin_prefetch((const void*)(Bt + (size_t)(n0 + lane)      * K + k0 + 32), 0, 3);
      __builtin_prefetch((const void*)(Bt + (size_t)(n0 + 32 + lane) * K + k0 + 32), 0, 3);
    }
    v16h a0 = load_frag_a(A + (size_t)m0        * K + k0, K);
    v16h a1 = load_frag_a(A + (size_t)(m0 + 16) * K + k0, K);
    v16h b[4];
#pragma unroll
    for (int j = 0; j < 4; ++j)
      b[j] = load_frag_b(Bt + (size_t)(n0 + j * 16) * K + k0, K);
#pragma unroll
    for (int j = 0; j < 4; ++j) {
      acc[0][j] = wmma_f16(a0, b[j], acc[0][j]);
      acc[1][j] = wmma_f16(a1, b[j], acc[1][j]);
    }
  }

#pragma unroll
  for (int i = 0; i < 2; ++i) {
#pragma unroll
    for (int j = 0; j < 4; ++j) {
#pragma unroll
      for (int v = 0; v < 8; ++v) {
        const int m = m0 + i * 16 + v + hlf * 8;   // C layout: vgpr v, half -> row
        const int n = n0 + j * 16 + col;           // lane&15 -> column
        const float val = acc[i][j][v];
        if constexpr (EPI == EPI_F16) {
          ((_Float16*)Out)[(size_t)m * Nd + n] = (_Float16)val;
        } else if constexpr (EPI == EPI_VT) {
          // v-projection stored head-transposed: (B,H,CH,L); L=1024, H=16, CH=64
          const int bi = m >> 10, l = m & 1023;
          const int hh = n >> 6,  c = n & 63;
          ((_Float16*)Out)[(((size_t)bi * 16 + hh) * 64 + c) * 1024 + l] = (_Float16)val;
        } else {
          ((float*)Out)[(size_t)m * Nd + n] = val + bias[n];
        }
      }
    }
  }
}

// ---------------------------------------------------------------------------
// Flash-style attention. 256 threads = 8 waves; each wave: 16 query rows,
// streams all L=1024 keys (32/iter) with online softmax. grid=(N/128, H, B).
// Per iter: 4 WMMA (scores) + 1 WMMA (row-sum vs ones) + 4 WMMA (P*V).
// ---------------------------------------------------------------------------
__global__ __launch_bounds__(256)
void attn_kernel(const _Float16* __restrict__ q16, const _Float16* __restrict__ k16,
                 const _Float16* __restrict__ vT,  const int* __restrict__ pad,
                 _Float16* __restrict__ o16) {
  constexpr int Lk = 1024, D = 1024, CH = 64, Nq = 1024, H = 16;
  const float NEG = -3.0e38f;

  const int w    = threadIdx.x >> 5;
  const int lane = threadIdx.x & 31;
  const int hlf  = lane >> 4, col = lane & 15;
  const int b = blockIdx.z, h = blockIdx.y;
  const int q0 = blockIdx.x * 128 + w * 16;

  __shared__ _Float16 pbuf[8][16][40];   // per-wave C-layout -> A-layout bounce

  const _Float16* qbase = q16 + ((size_t)(b * Nq + q0)) * D + h * CH;
  const _Float16* kbase = k16 + (size_t)b * Lk * D + h * CH;
  const _Float16* vbase = vT + ((size_t)(b * H + h)) * CH * Lk;
  const int* pmrow = pad + b * Lk;

  v16h qa0 = load_frag_a(qbase, D);        // channels [0,32)
  v16h qa1 = load_frag_a(qbase + 32, D);   // channels [32,64)
  // fold softmax scale (CH^-0.5 = 0.125) into q once -> v_pk_mul_f16
#pragma unroll
  for (int i = 0; i < 16; ++i) {
    qa0[i] = qa0[i] * (_Float16)0.125f;
    qa1[i] = qa1[i] * (_Float16)0.125f;
  }
  // all-ones B fragment: rowsum(P) = P x ones via one extra WMMA
  v16h onesb;
#pragma unroll
  for (int i = 0; i < 16; ++i) onesb[i] = (_Float16)1.0f;

  v8f oacc[4];
#pragma unroll
  for (int ct = 0; ct < 4; ++ct) oacc[ct] = (v8f){};
  float mrun[8], lrun[8];
#pragma unroll
  for (int v = 0; v < 8; ++v) { mrun[v] = -__builtin_inff(); lrun[v] = 0.0f; }

  for (int l0 = 0; l0 < Lk; l0 += 32) {
    // scores for two 16-key tiles, K = 64 channels (two WMMA steps each)
    v16h kb00 = load_frag_b(kbase + (size_t)l0 * D, D);
    v16h kb01 = load_frag_b(kbase + (size_t)l0 * D + 32, D);
    v16h kb10 = load_frag_b(kbase + (size_t)(l0 + 16) * D, D);
    v16h kb11 = load_frag_b(kbase + (size_t)(l0 + 16) * D + 32, D);
    v8f S0 = (v8f){};  S0 = wmma_f16(qa0, kb00, S0);  S0 = wmma_f16(qa1, kb01, S0);
    v8f S1 = (v8f){};  S1 = wmma_f16(qa0, kb10, S1);  S1 = wmma_f16(qa1, kb11, S1);

    const bool msk0 = pmrow[l0 + col] != 0;
    const bool msk1 = pmrow[l0 + 16 + col] != 0;

    float alpha[8];
#pragma unroll
    for (int v = 0; v < 8; ++v) {
      float s0 = msk0 ? NEG : S0[v];      // scale already folded into q
      float s1 = msk1 ? NEG : S1[v];
      float mx = dpp_bcast_max16(fmaxf(s0, s1));   // row max over 16 lanes
      float mnew = fmaxf(mrun[v], mx);
      alpha[v] = __expf(mrun[v] - mnew);
      mrun[v] = mnew;
      // P in C-layout -> LDS (A-layout bounce for the P*V / rowsum WMMAs)
      pbuf[w][v + hlf * 8][col]      = (_Float16)__expf(s0 - mnew);
      pbuf[w][v + hlf * 8][16 + col] = (_Float16)__expf(s1 - mnew);
    }
#pragma unroll
    for (int ct = 0; ct < 4; ++ct)
#pragma unroll
      for (int v = 0; v < 8; ++v) oacc[ct][v] *= alpha[v];

    __builtin_amdgcn_wave_barrier();
    v16h pf = load_frag_a(&pbuf[w][0][0], 40);

    // row-sum of P via ones-WMMA: every lane gets its rows' sums broadcast
    v8f ls = (v8f){};
    ls = wmma_f16(pf, onesb, ls);
#pragma unroll
    for (int v = 0; v < 8; ++v) lrun[v] = lrun[v] * alpha[v] + ls[v];

#pragma unroll
    for (int ct = 0; ct < 4; ++ct) {
      v16h vf = load_frag_b(vbase + (size_t)(ct * 16) * Lk + l0, Lk);
      oacc[ct] = wmma_f16(pf, vf, oacc[ct]);
    }
    __builtin_amdgcn_wave_barrier();
  }

  float inv[8];
#pragma unroll
  for (int v = 0; v < 8; ++v) inv[v] = 1.0f / lrun[v];
  _Float16* obase = o16 + ((size_t)(b * Nq + q0)) * D + h * CH;
#pragma unroll
  for (int ct = 0; ct < 4; ++ct)
#pragma unroll
    for (int v = 0; v < 8; ++v)
      obase[(size_t)(v + hlf * 8) * D + ct * 16 + col] =
          (_Float16)(oacc[ct][v] * inv[v]);
}

// ---------------------------------------------------------------------------
extern "C" void kernel_launch(void* const* d_in, const int* in_sizes, int n_in,
                              void* d_out, int out_size, void* d_ws, size_t ws_size,
                              hipStream_t stream) {
  const float* x_q  = (const float*)d_in[0];
  const float* x_kv = (const float*)d_in[1];
  const int*   pad  = (const int*)d_in[2];
  const float* Wq   = (const float*)d_in[3];
  const float* Wk   = (const float*)d_in[4];
  const float* Wv   = (const float*)d_in[5];
  const float* Wo   = (const float*)d_in[6];
  const float* bo   = (const float*)d_in[7];
  float* out = (float*)d_out;

  constexpr int Bq = 4, Nq = 1024, D = 1024;
  constexpr size_t MT = (size_t)Bq * Nq;     // 4096 token rows
  constexpr size_t TOK_BYTES = MT * D * sizeof(_Float16);   // 8 MiB
  constexpr size_t W_BYTES   = (size_t)D * D * sizeof(_Float16); // 2 MiB

  char* ws = (char*)d_ws;
  auto carve = [&](size_t bytes) { void* p = ws; ws += (bytes + 255) & ~(size_t)255; return p; };
  _Float16* xq16  = (_Float16*)carve(TOK_BYTES);
  _Float16* xkv16 = (_Float16*)carve(TOK_BYTES);
  _Float16* WqT   = (_Float16*)carve(W_BYTES);
  _Float16* WkT   = (_Float16*)carve(W_BYTES);
  _Float16* WvT   = (_Float16*)carve(W_BYTES);
  _Float16* WoT   = (_Float16*)carve(W_BYTES);
  _Float16* q16   = (_Float16*)carve(TOK_BYTES);
  _Float16* k16   = (_Float16*)carve(TOK_BYTES);
  _Float16* vT16  = (_Float16*)carve(TOK_BYTES);
  _Float16* o16   = (_Float16*)carve(TOK_BYTES);
  (void)ws_size; (void)in_sizes; (void)n_in; (void)out_size;

  // 1) activation converts
  cvt_f32_f16_kernel<<<2048, 256, 0, stream>>>(x_q,  xq16,  (int)(MT * D));
  cvt_f32_f16_kernel<<<2048, 256, 0, stream>>>(x_kv, xkv16, (int)(MT * D));
  // 2) weight transpose-converts (KxN f32 -> NxK f16)
  const int wBlocks = (D * D) / 256;
  cvt_transpose_kernel<<<wBlocks, 256, 0, stream>>>(Wq, WqT, D, D);
  cvt_transpose_kernel<<<wBlocks, 256, 0, stream>>>(Wk, WkT, D, D);
  cvt_transpose_kernel<<<wBlocks, 256, 0, stream>>>(Wv, WvT, D, D);
  cvt_transpose_kernel<<<wBlocks, 256, 0, stream>>>(Wo, WoT, D, D);
  // 3) projections
  dim3 gg(D / 128, MT / 128);   // (8, 32)
  gemm_wmma_kernel<EPI_F16>     <<<gg, 256, 0, stream>>>(xq16,  WqT, q16,  nullptr, (int)MT, D, D);
  gemm_wmma_kernel<EPI_F16>     <<<gg, 256, 0, stream>>>(xkv16, WkT, k16,  nullptr, (int)MT, D, D);
  gemm_wmma_kernel<EPI_VT>      <<<gg, 256, 0, stream>>>(xkv16, WvT, vT16, nullptr, (int)MT, D, D);
  // 4) attention (grid: N/128 row-blocks x H x B)
  attn_kernel<<<dim3(Nq / 128, 16, Bq), 256, 0, stream>>>(q16, k16, vT16, pad, o16);
  // 5) output projection + bias, f32 out
  gemm_wmma_kernel<EPI_F32_BIAS><<<gg, 256, 0, stream>>>(o16, WoT, out, bo, (int)MT, D, D);
}
